// ResourceAwareAttention_29429115912811
// MI455X (gfx1250) — compile-verified, matrix operand order checked by reading, not run on hardware
//
#include <hip/hip_runtime.h>
#include <hip/hip_bf16.h>
#include <math.h>

// ---------------------------------------------------------------------------
// Problem constants
// ---------------------------------------------------------------------------
#define BATCH   2
#define SEQ     2048
#define DMODEL  1024
#define NHEADS  16
#define DK      64
#define RTYPES  8
#define MROWS   (BATCH * SEQ)          // 4096 total token rows

typedef __attribute__((ext_vector_type(16))) __bf16 v16bf;
typedef __attribute__((ext_vector_type(8)))  float  v8f;
typedef __attribute__((ext_vector_type(4)))  unsigned int v4u;
typedef __attribute__((ext_vector_type(8)))  int v8i_t;
typedef __attribute__((ext_vector_type(4)))  int v4i_t;

union BFrag {
    v16bf v;
    uint4 u[2];    // 2 x 16B = 16 bf16
};

static __device__ __forceinline__ v8f wmma_bf16(const BFrag& a, const BFrag& b, v8f c) {
    return __builtin_amdgcn_wmma_f32_16x16x32_bf16(false, a.v, false, b.v,
                                                   (short)0, c, false, false);
}

#if __has_builtin(__builtin_amdgcn_tensor_load_to_lds) && __has_builtin(__builtin_amdgcn_s_wait_tensorcnt)
#define HAVE_TDM 1
// Issue a 2D TDM tile load (global -> LDS). Dimensions/strides in elements of
// 2 bytes (data_size=1 => 2B). D# bit layout per CDNA5 ISA §8.3/8.4.
// This toolchain's builtin is the 6-arg form:
//   (uint32x4 g0, int32x8 g1, int32x4 g2, int32x4 g3, int32x8 pad, i32 cpol)
static __device__ __forceinline__ void tdm_load_2d(unsigned lds_addr,
                                                   const void* gptr,
                                                   unsigned tensor_d0,
                                                   unsigned tensor_d1,
                                                   unsigned tile_d0,
                                                   unsigned tile_d1,
                                                   unsigned long long stride0)
{
    const unsigned long long ga = (unsigned long long)(uintptr_t)gptr;
    v4u g0;
    g0[0] = 1u;                                               // count=1 (valid user D#)
    g0[1] = lds_addr;                                         // LDS byte address
    g0[2] = (unsigned)(ga & 0xFFFFFFFFu);                     // global_addr[31:0]
    g0[3] = (unsigned)((ga >> 32) & 0x01FFFFFFu) | 0x80000000u; // addr[56:32] | type=2
    v8i_t g1;
    g1[0] = (int)(1u << 16);                                  // wg_mask=0, data_size=2B
    g1[1] = (int)((tensor_d0 & 0xFFFFu) << 16);               // tensor_dim0[15:0]
    g1[2] = (int)((tensor_d0 >> 16) | ((tensor_d1 & 0xFFFFu) << 16));
    g1[3] = (int)(((tensor_d1 >> 16) & 0xFFFFu) | (tile_d0 << 16));
    g1[4] = (int)(tile_d1 & 0xFFFFu);                         // tile_dim1, tile_dim2=0
    g1[5] = (int)(unsigned)(stride0 & 0xFFFFFFFFull);         // dim0_stride[31:0]
    g1[6] = (int)(unsigned)((stride0 >> 32) & 0xFFFFull);     // dim0_stride[47:32]
    g1[7] = 0;                                                // dim1_stride (unused, 2D)
    const v4i_t z4 = {0, 0, 0, 0};
    const v8i_t z8 = {0, 0, 0, 0, 0, 0, 0, 0};
    __builtin_amdgcn_tensor_load_to_lds(g0, g1, z4, z4, z8, 0);
}
#endif

// ---------------------------------------------------------------------------
// f32 -> bf16 conversion
// ---------------------------------------------------------------------------
__global__ __launch_bounds__(256)
void cvt_f32_bf16_kernel(const float* __restrict__ src, __bf16* __restrict__ dst, int n) {
    int i = blockIdx.x * 256 + threadIdx.x;
    if (i < n) dst[i] = (__bf16)src[i];
}

// ---------------------------------------------------------------------------
// GEMM: out[m,n] = sum_k A[m,k] * W[n,k] + bias[n]
//   A: [M, 1024] bf16 row-major, W: [1024, 1024] bf16 row-major (B[k,n]=W[n,k])
// mode 0: f32 row-major [M,1024] (final projection)
// mode 1: bf16 head-split [B,H,S,64]   (Q, K)
// mode 2: bf16 head-split transposed [B,H,64,S]  (V)
// Block = 256 threads = 8 waves; wave tile 32(M)x64(N) => 2x4 C frags.
// Waves: 4 along M, 2 along N => block tile 128x128.  grid = (1024/128, M/128)
// ---------------------------------------------------------------------------
__global__ __launch_bounds__(256)
void gemm_bf16_kernel(const __bf16* __restrict__ A,
                      const __bf16* __restrict__ W,
                      const float*  __restrict__ bias,
                      void* __restrict__ out,
                      int mode)
{
    const int lane = threadIdx.x & 31;
    const int wave = threadIdx.x >> 5;
    const int hf   = lane >> 4;      // half-wave select
    const int lr   = lane & 15;

    const int wm = wave & 3;         // 4 waves along M
    const int wn = wave >> 2;        // 2 waves along N
    const int m_base = blockIdx.y * 128 + wm * 32;
    const int n_base = blockIdx.x * 128 + wn * 64;

    const __bf16* arow0 = A + (size_t)(m_base + lr) * DMODEL;
    const __bf16* arow1 = arow0 + (size_t)16 * DMODEL;
    const __bf16* brow[4];
    #pragma unroll
    for (int ni = 0; ni < 4; ++ni)
        brow[ni] = W + (size_t)(n_base + ni * 16 + lr) * DMODEL;

    v8f c[2][4] = {};

    for (int k0 = 0; k0 < DMODEL; k0 += 32) {
        const int klo = k0 + hf * 8;
        const int khi = k0 + 16 + hf * 8;
        BFrag a0, a1, bb[4];
        a0.u[0] = *(const uint4*)(arow0 + klo);  a0.u[1] = *(const uint4*)(arow0 + khi);
        a1.u[0] = *(const uint4*)(arow1 + klo);  a1.u[1] = *(const uint4*)(arow1 + khi);
        #pragma unroll
        for (int ni = 0; ni < 4; ++ni) {
            bb[ni].u[0] = *(const uint4*)(brow[ni] + klo);
            bb[ni].u[1] = *(const uint4*)(brow[ni] + khi);
        }
        #pragma unroll
        for (int ni = 0; ni < 4; ++ni) {
            c[0][ni] = wmma_bf16(a0, bb[ni], c[0][ni]);
            c[1][ni] = wmma_bf16(a1, bb[ni], c[1][ni]);
        }
    }

    #pragma unroll
    for (int mi = 0; mi < 2; ++mi) {
        #pragma unroll
        for (int ni = 0; ni < 4; ++ni) {
            const int n  = n_base + ni * 16 + lr;
            const float bv = bias[n];
            #pragma unroll
            for (int v = 0; v < 8; ++v) {
                const int m = m_base + mi * 16 + v + 8 * hf;
                const float val = c[mi][ni][v] + bv;
                if (mode == 0) {
                    ((float*)out)[(size_t)m * DMODEL + n] = val;
                } else {
                    const int b  = m >> 11;          // / SEQ
                    const int s  = m & (SEQ - 1);
                    const int h  = n >> 6;           // / DK
                    const int dk = n & (DK - 1);
                    if (mode == 1) {
                        ((__bf16*)out)[((((size_t)b * NHEADS + h) * SEQ + s) << 6) + dk] = (__bf16)val;
                    } else {
                        ((__bf16*)out)[(((size_t)b * NHEADS + h) * DK + dk) * SEQ + s] = (__bf16)val;
                    }
                }
            }
        }
    }
}

// ---------------------------------------------------------------------------
// Flash attention with resource bias.
//   Q,K : [B*H, S, 64] bf16     Vt : [B*H, 64, S] bf16
//   Aout: [B, S, 1024] bf16 (merged heads)
// One block = one (b,h) and a 128-row Q chunk; 8 waves each own 16 Q rows.
// K/V slabs are staged into LDS by the Tensor Data Mover with DOUBLE
// BUFFERING: wave 0 issues the next slab's descriptors before waiting
// (tensorcnt <= 2 => current pair landed, next pair in flight during compute).
// Falls back to cooperative vector loads if the TDM builtin is unavailable.
// grid = (S/128, B*H) = (16, 32)
// ---------------------------------------------------------------------------
__global__ __launch_bounds__(256)
void flash_attn_kernel(const __bf16* __restrict__ Q,
                       const __bf16* __restrict__ Km,
                       const __bf16* __restrict__ Vt,
                       const int*    __restrict__ rmask,
                       const int*    __restrict__ amask,
                       const float*  __restrict__ rbias,
                       __bf16* __restrict__ Aout)
{
    __shared__ __align__(16) __bf16 kslab[2][32][DK];  // [buf][j_local][d]  2x4 KB
    __shared__ __align__(16) __bf16 vslab[2][DK][32];  // [buf][d][j_local]  2x4 KB
    __shared__ __align__(16) __bf16 pbuf[8][16][32];   // per-wave P tile 8 KB
    __shared__ float bbuf[RTYPES * RTYPES];            // this head's bias table

    const int tid  = threadIdx.x;
    const int lane = tid & 31;
    const int wave = tid >> 5;
    const int hf   = lane >> 4;
    const int lr   = lane & 15;

    const int bh = blockIdx.y;                         // 0 .. 31
    const int b  = bh >> 4;
    const int h  = bh & 15;
    const int i0 = blockIdx.x * 128 + wave * 16;       // this wave's Q rows

    if (tid < RTYPES * RTYPES) bbuf[tid] = rbias[h * RTYPES * RTYPES + tid];

    const size_t base = (size_t)bh * (SEQ * DK);

    // Q A-fragments: 16 x 64 = two 16x32 frags (direct from global, reused all loop)
    BFrag qa[2];
    const __bf16* qrow = Q + base + (size_t)(i0 + lr) * DK;
    qa[0].u[0] = *(const uint4*)(qrow + hf * 8);
    qa[0].u[1] = *(const uint4*)(qrow + 16 + hf * 8);
    qa[1].u[0] = *(const uint4*)(qrow + 32 + hf * 8);
    qa[1].u[1] = *(const uint4*)(qrow + 48 + hf * 8);

    // resource ids of the 8 rows this lane accumulates (rows v + 8*hf)
    int ri[8];
    #pragma unroll
    for (int v = 0; v < 8; ++v) ri[v] = rmask[b * SEQ + i0 + v + 8 * hf];

    float mrow[8], lrow[8];
    #pragma unroll
    for (int v = 0; v < 8; ++v) { mrow[v] = -INFINITY; lrow[v] = 0.0f; }
    v8f acc[4] = {};

#if defined(HAVE_TDM)
    // ---- pipeline prologue: DMA slab pair for jt=0 into buffer 0
    if (wave == 0) {
        tdm_load_2d((unsigned)(uintptr_t)&kslab[0][0][0],
                    Km + base, DK, SEQ, DK, 32, DK);
        tdm_load_2d((unsigned)(uintptr_t)&vslab[0][0][0],
                    Vt + base, SEQ, DK, 32, DK, SEQ);
    }
#else
    // cooperative slab-load addressing (per thread, one b128 each for K and V)
    const int krow_l = tid >> 3;            // 0..31  (j_local)
    const int kcol_l = (tid & 7) * 8;       // 0..56  (d)
    const int vrow_l = tid >> 2;            // 0..63  (d)
    const int vcol_l = (tid & 3) * 8;       // 0..24  (j_local)
#endif

    const float scale = 0.125f;             // 1/sqrt(64)

    for (int jt = 0; jt < SEQ; jt += 32) {
        const int cur = (jt >> 5) & 1;
#if defined(HAVE_TDM)
        if (wave == 0) {
            const int nxt = cur ^ 1;
            if (jt + 32 < SEQ) {
                // Prefetch next slab pair into the other buffer (safe: that
                // buffer's readers all finished before the previous iteration's pbuf
                // barrier), then wait for the CURRENT pair only.
                tdm_load_2d((unsigned)(uintptr_t)&kslab[nxt][0][0],
                            Km + base + (size_t)(jt + 32) * DK,
                            DK, SEQ, DK, 32, DK);
                tdm_load_2d((unsigned)(uintptr_t)&vslab[nxt][0][0],
                            Vt + base + jt + 32,
                            SEQ, DK, 32, DK, SEQ);
                __builtin_amdgcn_s_wait_tensorcnt(2);
            } else {
                __builtin_amdgcn_s_wait_tensorcnt(0);
            }
        }
#else
        *(uint4*)&kslab[cur][krow_l][kcol_l] =
            *(const uint4*)(Km + base + (size_t)(jt + krow_l) * DK + kcol_l);
        *(uint4*)&vslab[cur][vrow_l][vcol_l] =
            *(const uint4*)(Vt + base + (size_t)vrow_l * SEQ + jt + vcol_l);
#endif
        __syncthreads();

        // ---- K B-frags (2 col groups x 2 d-chunks) and V B-frags from LDS
        BFrag kf[2][2], vf[4];
        #pragma unroll
        for (int cj = 0; cj < 2; ++cj) {
            const int n = cj * 16 + lr;
            #pragma unroll
            for (int kc = 0; kc < 2; ++kc) {
                kf[cj][kc].u[0] = *(const uint4*)(&kslab[cur][n][kc * 32 + hf * 8]);
                kf[cj][kc].u[1] = *(const uint4*)(&kslab[cur][n][kc * 32 + 16 + hf * 8]);
            }
        }
        #pragma unroll
        for (int cd = 0; cd < 4; ++cd) {
            const int d = cd * 16 + lr;
            vf[cd].u[0] = *(const uint4*)(&vslab[cur][d][hf * 8]);
            vf[cd].u[1] = *(const uint4*)(&vslab[cur][d][16 + hf * 8]);
        }

        // ---- scores: Q @ K^T over the 32-column j slab
        v8f sc[2] = {};
        #pragma unroll
        for (int cj = 0; cj < 2; ++cj) {
            sc[cj] = wmma_bf16(qa[0], kf[cj][0], sc[cj]);
            sc[cj] = wmma_bf16(qa[1], kf[cj][1], sc[cj]);
        }

        // ---- scale + resource bias + attention mask
        #pragma unroll
        for (int cj = 0; cj < 2; ++cj) {
            const int j  = jt + cj * 16 + lr;
            const int rj = rmask[b * SEQ + j];
            const int am = amask[b * SEQ + j];
            #pragma unroll
            for (int v = 0; v < 8; ++v) {
                float s = sc[cj][v] * scale + bbuf[ri[v] * RTYPES + rj];
                if (am == 0) s = -1.0e9f;
                sc[cj][v] = s;
            }
        }

        // ---- online softmax (row reduction over the 16-lane half groups)
        float ef[8];
        #pragma unroll
        for (int v = 0; v < 8; ++v) {
            float mx = fmaxf(sc[0][v], sc[1][v]);
            mx = fmaxf(mx, __shfl_xor(mx, 1, 32));
            mx = fmaxf(mx, __shfl_xor(mx, 2, 32));
            mx = fmaxf(mx, __shfl_xor(mx, 4, 32));
            mx = fmaxf(mx, __shfl_xor(mx, 8, 32));
            const float mnew = fmaxf(mrow[v], mx);
            const float p0 = __expf(sc[0][v] - mnew);
            const float p1 = __expf(sc[1][v] - mnew);
            sc[0][v] = p0; sc[1][v] = p1;
            float rs = p0 + p1;
            rs += __shfl_xor(rs, 1, 32);
            rs += __shfl_xor(rs, 2, 32);
            rs += __shfl_xor(rs, 4, 32);
            rs += __shfl_xor(rs, 8, 32);
            ef[v]   = __expf(mrow[v] - mnew);
            lrow[v] = lrow[v] * ef[v] + rs;
            mrow[v] = mnew;
        }

        // ---- relayout P (C-frag) -> A-frag via LDS, in bf16
        #pragma unroll
        for (int v = 0; v < 8; ++v) {
            const int row = v + 8 * hf;
            pbuf[wave][row][lr]      = (__bf16)sc[0][v];
            pbuf[wave][row][lr + 16] = (__bf16)sc[1][v];
        }
        // This barrier publishes pbuf AND, since all kslab/vslab reads (kf/vf)
        // happened above it, licenses overwriting this buffer next time around.
        __syncthreads();
        BFrag pa;
        pa.u[0] = *(const uint4*)(&pbuf[wave][lr][hf * 8]);
        pa.u[1] = *(const uint4*)(&pbuf[wave][lr][16 + hf * 8]);

        // ---- rescale accumulators and add P @ V slab (V frags already in regs)
        #pragma unroll
        for (int cd = 0; cd < 4; ++cd) {
            #pragma unroll
            for (int v = 0; v < 8; ++v) acc[cd][v] *= ef[v];
            acc[cd] = wmma_bf16(pa, vf[cd], acc[cd]);
        }
    }

    // ---- normalize and store merged-head output (bf16 for the final GEMM)
    #pragma unroll
    for (int cd = 0; cd < 4; ++cd) {
        const int d = cd * 16 + lr;
        #pragma unroll
        for (int v = 0; v < 8; ++v) {
            const int i = i0 + v + 8 * hf;
            const float o = acc[cd][v] / lrow[v];
            Aout[((size_t)(b * SEQ + i)) * DMODEL + h * DK + d] = (__bf16)o;
        }
    }
}

// ---------------------------------------------------------------------------
// Host-side launch
// ---------------------------------------------------------------------------
extern "C" void kernel_launch(void* const* d_in, const int* in_sizes, int n_in,
                              void* d_out, int out_size, void* d_ws, size_t ws_size,
                              hipStream_t stream) {
    const float* x     = (const float*)d_in[0];
    const float* Wq    = (const float*)d_in[1];
    const float* bq    = (const float*)d_in[2];
    const float* Wk    = (const float*)d_in[3];
    const float* bk    = (const float*)d_in[4];
    const float* Wv    = (const float*)d_in[5];
    const float* bv    = (const float*)d_in[6];
    const float* Wo    = (const float*)d_in[7];
    const float* bo    = (const float*)d_in[8];
    const float* rbias = (const float*)d_in[9];
    const int*   rmask = (const int*)d_in[10];
    const int*   amask = (const int*)d_in[11];

    // workspace carve-up (bytes), all 256B aligned
    char* ws = (char*)d_ws;
    const size_t SZ_X  = (size_t)MROWS  * DMODEL * sizeof(__bf16);   // 8 MiB
    const size_t SZ_W  = (size_t)DMODEL * DMODEL * sizeof(__bf16);   // 2 MiB
    __bf16* xb  = (__bf16*)(ws);                  ws += SZ_X;
    __bf16* wqb = (__bf16*)(ws);                  ws += SZ_W;
    __bf16* wkb = (__bf16*)(ws);                  ws += SZ_W;
    __bf16* wvb = (__bf16*)(ws);                  ws += SZ_W;
    __bf16* wob = (__bf16*)(ws);                  ws += SZ_W;
    __bf16* qb  = (__bf16*)(ws);                  ws += SZ_X;        // [B,H,S,64]
    __bf16* kb  = (__bf16*)(ws);                  ws += SZ_X;        // [B,H,S,64]
    __bf16* vt  = (__bf16*)(ws);                  ws += SZ_X;        // [B,H,64,S]
    __bf16* aob = (__bf16*)(ws);                  ws += SZ_X;        // [B,S,1024]

    const int NX = MROWS * DMODEL;     // 4194304
    const int NW = DMODEL * DMODEL;    // 1048576

    cvt_f32_bf16_kernel<<<(NX + 255) / 256, 256, 0, stream>>>(x,  xb,  NX);
    cvt_f32_bf16_kernel<<<(NW + 255) / 256, 256, 0, stream>>>(Wq, wqb, NW);
    cvt_f32_bf16_kernel<<<(NW + 255) / 256, 256, 0, stream>>>(Wk, wkb, NW);
    cvt_f32_bf16_kernel<<<(NW + 255) / 256, 256, 0, stream>>>(Wv, wvb, NW);
    cvt_f32_bf16_kernel<<<(NW + 255) / 256, 256, 0, stream>>>(Wo, wob, NW);

    dim3 ggrid(DMODEL / 128, MROWS / 128);   // (8, 32)
    gemm_bf16_kernel<<<ggrid, 256, 0, stream>>>(xb, wqb, bq, qb, 1);
    gemm_bf16_kernel<<<ggrid, 256, 0, stream>>>(xb, wkb, bk, kb, 1);
    gemm_bf16_kernel<<<ggrid, 256, 0, stream>>>(xb, wvb, bv, vt, 2);

    dim3 agrid(SEQ / 128, BATCH * NHEADS);   // (16, 32)
    flash_attn_kernel<<<agrid, 256, 0, stream>>>(qb, kb, vt, rmask, amask, rbias, aob);

    gemm_bf16_kernel<<<ggrid, 256, 0, stream>>>(aob, wob, bo, d_out, 0);
}
